// CoAttentionNetwork_68221260530117
// MI455X (gfx1250) — compile-verified, matrix operand order checked by reading, not run on hardware
//
#include <hip/hip_runtime.h>

typedef unsigned short u16;
typedef unsigned int   u32;
typedef __attribute__((ext_vector_type(16))) __bf16 v16bf;
typedef __attribute__((ext_vector_type(8)))  float  v8f;
typedef int v4i __attribute__((vector_size(16)));

#define AS1 __attribute__((address_space(1)))
#define AS3 __attribute__((address_space(3)))

#if defined(__has_builtin)
# if __has_builtin(__builtin_amdgcn_global_load_async_to_lds_b128)
#  define HAS_ASYNC_LDS 1
# endif
# if __has_builtin(__builtin_amdgcn_s_wait_asynccnt)
#  define HAS_WAIT_ASYNC 1
# endif
#endif
#ifndef HAS_ASYNC_LDS
# define HAS_ASYNC_LDS 0
#endif
#ifndef HAS_WAIT_ASYNC
# define HAS_WAIT_ASYNC 0
#endif

// ---------------------------------------------------------------- utilities

union Frag {
    u32   u[8];
    u16   s[16];
    v16bf v;
};

__device__ __forceinline__ u16 f2bf(float f) {
    u32 u = __float_as_uint(f);
    u32 r = u + 0x7FFFu + ((u >> 16) & 1u);   // round-to-nearest-even
    return (u16)(r >> 16);
}

// CDNA5 16-bit A/B fragment K mapping (ISA 7.12.2):
//  lanes 0-15 hold K in {0..7, 16..23}; lanes 16-31 hold K in {8..15, 24..31}
__device__ __forceinline__ int kpair(int r, int hl) {
    return (r < 4 ? 2 * r : 16 + 2 * (r - 4)) + 8 * hl;
}

__device__ __forceinline__ v8f bf16_wmma(Frag a, Frag b, v8f c) {
    return __builtin_amdgcn_wmma_f32_16x16x32_bf16(
        false, a.v, false, b.v, (short)0, c, false, false);
}

// 16-byte global -> LDS copy: async (ASYNCcnt-tracked) when available.
__device__ __forceinline__ void async_cp16(const u16* g, u16* l) {
#if HAS_ASYNC_LDS
    __builtin_amdgcn_global_load_async_to_lds_b128(
        (AS1 v4i*)(size_t)g,    // flat address == global address for global mem
        (AS3 v4i*)l,            // generic -> LDS addrspacecast
        0, 0);
#else
    *(uint4*)l = *(const uint4*)g;
#endif
}

__device__ __forceinline__ void wait_async0() {
#if HAS_WAIT_ASYNC
    __builtin_amdgcn_s_wait_asynccnt(0);
#elif HAS_ASYNC_LDS
    asm volatile("s_wait_asynccnt 0x0" ::: "memory");
#endif
}

// ---------------------------------------------------------------- converts

__global__ void cvt_bf16(const float* __restrict__ src, u16* __restrict__ dst, int n) {
    int i = blockIdx.x * blockDim.x + threadIdx.x;
    int stride = gridDim.x * blockDim.x;
    for (; i < n; i += stride) dst[i] = f2bf(src[i]);
}

// Transposing convert: src [K][N] f32 -> dst [N][K] bf16 (weights become
// N-major so GEMM B tiles are contiguous row-major copies -> async-copyable).
__global__ __launch_bounds__(256) void cvt_bf16_t(
    const float* __restrict__ src, u16* __restrict__ dst, int K, int N)
{
    __shared__ u16 t[32][33];
    const int n0 = blockIdx.x * 32, k0 = blockIdx.y * 32;
    #pragma unroll
    for (int i = threadIdx.y; i < 32; i += 8)
        t[i][threadIdx.x] = f2bf(src[(size_t)(k0 + i) * N + n0 + threadIdx.x]);
    __syncthreads();
    #pragma unroll
    for (int i = threadIdx.y; i < 32; i += 8)
        dst[(size_t)(n0 + i) * K + k0 + threadIdx.x] = t[threadIdx.x][i];
}

// ---------------------------------------------------------------- projection GEMM
// [4096 x 1024] (bf16) @ Wt [1024(N) x 1024(K)] (bf16) -> bf16
// 256 threads = 8 waves; tile 64(M) x 128(N); BK=32; double-buffered async LDS.

__global__ __launch_bounds__(256) void proj_gemm(
    const u16* __restrict__ xq, const u16* __restrict__ xc,
    const u16* __restrict__ wtq, const u16* __restrict__ wtc,
    u16* __restrict__ pq, u16* __restrict__ pc)
{
    const u16* x  = blockIdx.z ? xc  : xq;
    const u16* wt = blockIdx.z ? wtc : wtq;
    u16*       p  = blockIdx.z ? pc  : pq;

    // pad rows to 40 u16 (80B): keeps 16B alignment for B128 async chunks and
    // walks all 64 LDS banks on fragment reads.
    __shared__ u16 Ab[2][64][40];
    __shared__ u16 Bb[2][128][40];

    const int tid  = threadIdx.x;
    const int lane = tid & 31, wid = tid >> 5;
    const int hl   = (lane >> 4) & 1, mr = lane & 15;
    const int wm   = wid & 3, wn = wid >> 2;
    const int row0 = blockIdx.x * 64;
    const int col0 = blockIdx.y * 128;

    const int ar = tid >> 2;            // 0..63
    const int ac = (tid & 3) * 8;       // chunk of 8 bf16 = 16B

    v8f acc0 = {}, acc1 = {}, acc2 = {}, acc3 = {};

    auto stage = [&](int k0, int buf) {
        async_cp16(x  + (size_t)(row0 + ar)      * 1024 + k0 + ac, &Ab[buf][ar][ac]);
        async_cp16(wt + (size_t)(col0 + ar)      * 1024 + k0 + ac, &Bb[buf][ar][ac]);
        async_cp16(wt + (size_t)(col0 + 64 + ar) * 1024 + k0 + ac, &Bb[buf][64 + ar][ac]);
    };

    stage(0, 0);
    int buf = 0;
    for (int k0 = 0; k0 < 1024; k0 += 32) {
        wait_async0();
        __syncthreads();
        if (k0 + 32 < 1024) stage(k0 + 32, buf ^ 1);

        Frag a;
        #pragma unroll
        for (int r = 0; r < 8; r++)
            a.u[r] = *(const u32*)&Ab[buf][wm * 16 + mr][kpair(r, hl)];

        Frag b0, b1, b2, b3;
        #pragma unroll
        for (int r = 0; r < 8; r++) {
            int kk = kpair(r, hl);
            b0.u[r] = *(const u32*)&Bb[buf][wn * 64 +  0 + mr][kk];
            b1.u[r] = *(const u32*)&Bb[buf][wn * 64 + 16 + mr][kk];
            b2.u[r] = *(const u32*)&Bb[buf][wn * 64 + 32 + mr][kk];
            b3.u[r] = *(const u32*)&Bb[buf][wn * 64 + 48 + mr][kk];
        }
        acc0 = bf16_wmma(a, b0, acc0);
        acc1 = bf16_wmma(a, b1, acc1);
        acc2 = bf16_wmma(a, b2, acc2);
        acc3 = bf16_wmma(a, b3, acc3);
        buf ^= 1;
    }

    #pragma unroll
    for (int j = 0; j < 8; j++) {
        int row = row0 + wm * 16 + j + 8 * hl;   // C/D layout: reg j -> row j + 8*half
        int cb  = col0 + wn * 64 + mr;
        p[(size_t)row * 1024 + cb     ] = f2bf(acc0[j]);
        p[(size_t)row * 1024 + cb + 16] = f2bf(acc1[j]);
        p[(size_t)row * 1024 + cb + 32] = f2bf(acc2[j]);
        p[(size_t)row * 1024 + cb + 48] = f2bf(acc3[j]);
    }
}

// ---------------------------------------------------------------- flash attention
// One wave per (16 query rows, head, batch, direction). S=1024, DK=64.
// type 0: question_ = attn(q=proj_c, kv=proj_q, mask_q)
// type 1: context_  = attn(q=proj_q, kv=proj_c, mask_c)

__global__ __launch_bounds__(32) void attn_kernel(
    const u16* __restrict__ projq, const u16* __restrict__ projc,
    const int* __restrict__ mask_q, const int* __restrict__ mask_c,
    u16* __restrict__ attnq, u16* __restrict__ attnc)
{
    const int lane = threadIdx.x;
    const int hl = lane >> 4, mr = lane & 15;
    const int qb = blockIdx.x * 16;
    const int h  = blockIdx.y;
    const int b  = blockIdx.z >> 1, type = blockIdx.z & 1;

    const u16* Q     = type ? projq : projc;
    const u16* KV    = type ? projc : projq;
    const int* maskp = type ? mask_c : mask_q;
    u16*       out   = type ? attnc : attnq;

    Frag aq0, aq1;
    {
        const u16* qrow = Q + ((size_t)(b * 1024 + qb + mr)) * 1024 + h * 64;
        #pragma unroll
        for (int r = 0; r < 8; r++) {
            int k = kpair(r, hl);
            aq0.u[r] = *(const u32*)(qrow + k);
            aq1.u[r] = *(const u32*)(qrow + 32 + k);
        }
    }

    bool mneg[8];
    #pragma unroll
    for (int j = 0; j < 8; j++)
        mneg[j] = (maskp[b * 1024 + qb + j + 8 * hl] == 0);

    float mstat[8], lstat[8];
    #pragma unroll
    for (int j = 0; j < 8; j++) { mstat[j] = -1.0e30f; lstat[j] = 0.0f; }
    v8f o0 = {}, o1 = {}, o2 = {}, o3 = {};

    __shared__ u16 Pt[16][36];

    for (int kb = 0; kb < 1024; kb += 32) {
        Frag bk0, bk1, bk2, bk3;
        {
            const u16* kr0 = KV + ((size_t)(b * 1024 + kb      + mr)) * 1024 + h * 64;
            const u16* kr1 = KV + ((size_t)(b * 1024 + kb + 16 + mr)) * 1024 + h * 64;
            #pragma unroll
            for (int r = 0; r < 8; r++) {
                int k = kpair(r, hl);
                bk0.u[r] = *(const u32*)(kr0 + k);
                bk1.u[r] = *(const u32*)(kr0 + 32 + k);
                bk2.u[r] = *(const u32*)(kr1 + k);
                bk3.u[r] = *(const u32*)(kr1 + 32 + k);
            }
        }
        v8f s0 = {}, s1 = {};
        s0 = bf16_wmma(aq0, bk0, s0);
        s0 = bf16_wmma(aq1, bk1, s0);
        s1 = bf16_wmma(aq0, bk2, s1);
        s1 = bf16_wmma(aq1, bk3, s1);

        float p0[8], p1[8];
        #pragma unroll
        for (int j = 0; j < 8; j++) {
            float v0 = s0[j] * 0.125f;          // 1/sqrt(DK)
            float v1 = s1[j] * 0.125f;
            if (mneg[j]) { v0 = -65504.0f; v1 = -65504.0f; }
            float bm = fmaxf(v0, v1);
            #pragma unroll
            for (int d = 1; d < 16; d <<= 1) bm = fmaxf(bm, __shfl_xor(bm, d, 32));
            float mn = fmaxf(mstat[j], bm);
            float al = __expf(mstat[j] - mn);
            p0[j] = __expf(v0 - mn);
            p1[j] = __expf(v1 - mn);
            float rs = p0[j] + p1[j];
            #pragma unroll
            for (int d = 1; d < 16; d <<= 1) rs += __shfl_xor(rs, d, 32);
            lstat[j] = lstat[j] * al + rs;
            mstat[j] = mn;
            o0[j] *= al; o1[j] *= al; o2[j] *= al; o3[j] *= al;
        }

        #pragma unroll
        for (int j = 0; j < 8; j++) {
            Pt[j + 8 * hl][mr]      = f2bf(p0[j]);
            Pt[j + 8 * hl][16 + mr] = f2bf(p1[j]);
        }
        __syncthreads();
        Frag ap;
        #pragma unroll
        for (int r = 0; r < 8; r++)
            ap.u[r] = *(const u32*)&Pt[mr][kpair(r, hl)];

        Frag bv0, bv1, bv2, bv3;
        #pragma unroll
        for (int e = 0; e < 16; e++) {
            int k = kpair(e >> 1, hl) + (e & 1);
            const u16* vrow = KV + ((size_t)(b * 1024 + kb + k)) * 1024 + h * 64;
            bv0.s[e] = vrow[ 0 + mr];
            bv1.s[e] = vrow[16 + mr];
            bv2.s[e] = vrow[32 + mr];
            bv3.s[e] = vrow[48 + mr];
        }
        o0 = bf16_wmma(ap, bv0, o0);
        o1 = bf16_wmma(ap, bv1, o1);
        o2 = bf16_wmma(ap, bv2, o2);
        o3 = bf16_wmma(ap, bv3, o3);
        __syncthreads();
    }

    #pragma unroll
    for (int j = 0; j < 8; j++) {
        float inv = 1.0f / lstat[j];
        int row = qb + j + 8 * hl;
        size_t base = ((size_t)(b * 1024 + row)) * 1024 + h * 64;
        out[base +  0 + mr] = f2bf(o0[j] * inv);
        out[base + 16 + mr] = f2bf(o1[j] * inv);
        out[base + 32 + mr] = f2bf(o2[j] * inv);
        out[base + 48 + mr] = f2bf(o3[j] * inv);
    }
}

// ---------------------------------------------------------------- FFN GEMM
// out = x + tanh(concat([x, x_attn]) @ W + b);  M=4096, N=1024, K=2048.
// Wt is [1024(N) x 2048(K)] bf16 (pre-transposed).

__global__ __launch_bounds__(256) void ffn_gemm(
    const u16* __restrict__ xqb, const u16* __restrict__ xcb,
    const u16* __restrict__ aqb, const u16* __restrict__ acb,
    const u16* __restrict__ w1t, const u16* __restrict__ w2t,
    const float* __restrict__ b1, const float* __restrict__ b2,
    const float* __restrict__ xq, const float* __restrict__ xc,
    float* __restrict__ outq, float* __restrict__ outc)
{
    const u16* src0  = blockIdx.z ? xcb : xqb;
    const u16* src1  = blockIdx.z ? acb : aqb;
    const u16* wt    = blockIdx.z ? w2t : w1t;
    const float* bias = blockIdx.z ? b2 : b1;
    const float* xf  = blockIdx.z ? xc : xq;
    float*       out = blockIdx.z ? outc : outq;

    __shared__ u16 Ab[2][64][40];
    __shared__ u16 Bb[2][128][40];

    const int tid  = threadIdx.x;
    const int lane = tid & 31, wid = tid >> 5;
    const int hl   = (lane >> 4) & 1, mr = lane & 15;
    const int wm   = wid & 3, wn = wid >> 2;
    const int row0 = blockIdx.x * 64;
    const int col0 = blockIdx.y * 128;

    const int ar = tid >> 2;
    const int ac = (tid & 3) * 8;

    v8f acc0 = {}, acc1 = {}, acc2 = {}, acc3 = {};

    auto stage = [&](int k0, int buf) {
        const u16* asrc = (k0 < 1024) ? src0 : src1;
        const int  ka   = k0 & 1023;
        async_cp16(asrc + (size_t)(row0 + ar)      * 1024 + ka + ac, &Ab[buf][ar][ac]);
        async_cp16(wt   + (size_t)(col0 + ar)      * 2048 + k0 + ac, &Bb[buf][ar][ac]);
        async_cp16(wt   + (size_t)(col0 + 64 + ar) * 2048 + k0 + ac, &Bb[buf][64 + ar][ac]);
    };

    stage(0, 0);
    int buf = 0;
    for (int k0 = 0; k0 < 2048; k0 += 32) {
        wait_async0();
        __syncthreads();
        if (k0 + 32 < 2048) stage(k0 + 32, buf ^ 1);

        Frag a;
        #pragma unroll
        for (int r = 0; r < 8; r++)
            a.u[r] = *(const u32*)&Ab[buf][wm * 16 + mr][kpair(r, hl)];

        Frag b0f, b1f, b2f, b3f;
        #pragma unroll
        for (int r = 0; r < 8; r++) {
            int kk = kpair(r, hl);
            b0f.u[r] = *(const u32*)&Bb[buf][wn * 64 +  0 + mr][kk];
            b1f.u[r] = *(const u32*)&Bb[buf][wn * 64 + 16 + mr][kk];
            b2f.u[r] = *(const u32*)&Bb[buf][wn * 64 + 32 + mr][kk];
            b3f.u[r] = *(const u32*)&Bb[buf][wn * 64 + 48 + mr][kk];
        }
        acc0 = bf16_wmma(a, b0f, acc0);
        acc1 = bf16_wmma(a, b1f, acc1);
        acc2 = bf16_wmma(a, b2f, acc2);
        acc3 = bf16_wmma(a, b3f, acc3);
        buf ^= 1;
    }

    const int colb = col0 + wn * 64 + mr;
    const float bb0 = bias[colb], bb1 = bias[colb + 16];
    const float bb2 = bias[colb + 32], bb3 = bias[colb + 48];
    #pragma unroll
    for (int j = 0; j < 8; j++) {
        size_t row = (size_t)(row0 + wm * 16 + j + 8 * hl);
        size_t i0 = row * 1024 + colb;
        out[i0     ] = xf[i0     ] + tanhf(acc0[j] + bb0);
        out[i0 + 16] = xf[i0 + 16] + tanhf(acc1[j] + bb1);
        out[i0 + 32] = xf[i0 + 32] + tanhf(acc2[j] + bb2);
        out[i0 + 48] = xf[i0 + 48] + tanhf(acc3[j] + bb3);
    }
}

// ---------------------------------------------------------------- launcher

extern "C" void kernel_launch(void* const* d_in, const int* in_sizes, int n_in,
                              void* d_out, int out_size, void* d_ws, size_t ws_size,
                              hipStream_t stream) {
    (void)in_sizes; (void)n_in; (void)out_size; (void)ws_size;

    const float* question = (const float*)d_in[0];
    const float* context  = (const float*)d_in[1];
    const int*   mask_q   = (const int*)d_in[2];
    const int*   mask_c   = (const int*)d_in[3];
    const float* Wq = (const float*)d_in[4];
    const float* Wc = (const float*)d_in[5];
    const float* W1 = (const float*)d_in[6];
    const float* b1 = (const float*)d_in[7];
    const float* W2 = (const float*)d_in[8];
    const float* b2 = (const float*)d_in[9];

    const size_t NX = (size_t)4 * 1024 * 1024;   // B*S*D
    u16* ws  = (u16*)d_ws;
    u16* xqb = ws;                u16* xcb = xqb + NX;
    u16* wqb = xcb + NX;          u16* wcb = wqb + (size_t)1024 * 1024;
    u16* w1b = wcb + (size_t)1024 * 1024;
    u16* w2b = w1b + (size_t)2048 * 1024;
    u16* pq  = w2b + (size_t)2048 * 1024;
    u16* pc  = pq + NX;
    u16* aq  = pc + NX;
    u16* ac  = aq + NX;

    float* outq = (float*)d_out;
    float* outc = outq + NX;

    cvt_bf16<<<512, 256, 0, stream>>>(question, xqb, (int)NX);
    cvt_bf16<<<512, 256, 0, stream>>>(context,  xcb, (int)NX);
    // weights converted AND transposed to [N][K] so GEMM tiles are contiguous
    cvt_bf16_t<<<dim3(32, 32), dim3(32, 8), 0, stream>>>(Wq, wqb, 1024, 1024);
    cvt_bf16_t<<<dim3(32, 32), dim3(32, 8), 0, stream>>>(Wc, wcb, 1024, 1024);
    cvt_bf16_t<<<dim3(32, 64), dim3(32, 8), 0, stream>>>(W1, w1b, 2048, 1024);
    cvt_bf16_t<<<dim3(32, 64), dim3(32, 8), 0, stream>>>(W2, w2b, 2048, 1024);

    proj_gemm<<<dim3(64, 8, 2), 256, 0, stream>>>(xqb, xcb, wqb, wcb, pq, pc);
    attn_kernel<<<dim3(64, 16, 8), 32, 0, stream>>>(pq, pc, mask_q, mask_c, aq, ac);
    ffn_gemm<<<dim3(64, 8, 2), 256, 0, stream>>>(xqb, xcb, aq, ac, w1b, w2b,
                                                 b1, b2, question, context, outq, outc);
}
